// GAT_66374424592903
// MI455X (gfx1250) — matrix operand order repaired and checked by reading, not compiled, not run on hardware
//
#include <hip/hip_runtime.h>

// ---------------------------------------------------------------------------
// GAT layer fused for MI455X (gfx1250, wave32, WMMA 16x16x32 f16->f32).
// Pipeline:
//   k_f32_to_f16   : x, W -> f16 workspace copies
//   k_pack_adj     : adj int32 [B,N,N] -> bitmask [B,N,N/32] (wave32 ballot)
//   k_hidden       : hidden = x @ W^T via v_wmma_f32_16x16x32_f16
//                    stores hidden f16 row-major AND transposed [B,H,COUT,N]
//   k_scores       : s1 = hidden . a[h,0], s2 = hidden . a[h,1]
//   k_attn         : flash-attention style fused leakyrelu+mask+softmax+P@V
//                    (online softmax, WMMA for P@V, LDS-staged V tiles),
//                    epilogue: /l + bias, ReLU, head-concat store.
// ---------------------------------------------------------------------------

typedef __attribute__((ext_vector_type(16))) _Float16 v16h;
typedef __attribute__((ext_vector_type(8)))  _Float16 v8h;
typedef __attribute__((ext_vector_type(8)))  float    v8f;

static constexpr int B    = 4;
static constexpr int N    = 2048;
static constexpr int CIN  = 256;
static constexpr int COUT = 64;
static constexpr int H    = 8;
static constexpr float LRELU_ALPHA = 0.2f;
static constexpr float NEG_BIG     = -1e12f;

// ---------------------------------------------------------------------------
__global__ void k_f32_to_f16(const float* __restrict__ src,
                             _Float16* __restrict__ dst, int n) {
  int i = blockIdx.x * blockDim.x + threadIdx.x;
  if (i < n) dst[i] = (_Float16)src[i];
}

// One wave32 per output word: ballot(adj>0) packs 32 ints into one u32.
__global__ void k_pack_adj(const int* __restrict__ adj,
                           unsigned* __restrict__ bits, int nwords) {
  int lane = threadIdx.x & 31;
  int w = blockIdx.x * (blockDim.x >> 5) + (threadIdx.x >> 5);
  if (w >= nwords) return;
  int v = adj[(size_t)w * 32 + lane];
  unsigned long long m = __ballot(v > 0);
  if (lane == 0) bits[w] = (unsigned)m;
}

// ---------------------------------------------------------------------------
// hidden[bh, m, o] = sum_i x[b, m, i] * W[h, o, i]
// Block: 256 thr = 8 waves, each wave computes 16 rows x 64 cols.
__global__ __launch_bounds__(256) void k_hidden(
    const _Float16* __restrict__ x16,   // [B, N, CIN]
    const _Float16* __restrict__ w16,   // [H, COUT, CIN]
    _Float16* __restrict__ hid,         // [B*H, N, COUT]
    _Float16* __restrict__ hidT) {      // [B*H, COUT, N]
  const int bh = blockIdx.x, b = bh >> 3, h = bh & 7;
  const int wave = threadIdx.x >> 5, lane = threadIdx.x & 31;
  const int n = lane & 15;
  const bool hi = lane >= 16;
  const int r0 = blockIdx.y * 128 + wave * 16;

  v8f acc[4] = {};
  const _Float16* xrow = x16 + (size_t)(b * N + r0 + n) * CIN;

#pragma unroll 4
  for (int kc = 0; kc < CIN; kc += 32) {
    // A-fragment: lane = row, K = kb+{0..15} contiguous, kb = hi?16:0
    v16h afrag;
    {
      const int kb = kc + (hi ? 16 : 0);
      v8h a0 = *(const v8h*)(xrow + kb);
      v8h a1 = *(const v8h*)(xrow + kb + 8);
#pragma unroll
      for (int i = 0; i < 8; ++i) { afrag[i] = a0[i]; afrag[8 + i] = a1[i]; }
    }
#pragma unroll
    for (int j = 0; j < 4; ++j) {
      // B-fragment: lane = col o, K = kc + (hi?16:0) .. +15 contiguous
      const int o = j * 16 + n;
      const v16h bfrag =
          *(const v16h*)(w16 + (size_t)(h * COUT + o) * CIN + kc + (hi ? 16 : 0));
      acc[j] = __builtin_amdgcn_wmma_f32_16x16x32_f16(
          false, afrag, false, bfrag, (short)0, acc[j], false, false);
    }
  }

  // C layout: lane<16 -> (M=r, N=lane); lane>=16 -> (M=8+r, N=lane-16)
#pragma unroll
  for (int r = 0; r < 8; ++r) {
    const int m = r0 + (hi ? 8 + r : r);
#pragma unroll
    for (int j = 0; j < 4; ++j) {
      const int o = j * 16 + n;
      _Float16 v = (_Float16)acc[j][r];
      hid[(size_t)(bh * N + m) * COUT + o] = v;
      hidT[(size_t)(bh * COUT + o) * N + m] = v;
    }
  }
}

// ---------------------------------------------------------------------------
__global__ void k_scores(const _Float16* __restrict__ hid,
                         const float* __restrict__ a,  // [H,2,COUT]
                         float* __restrict__ s1, float* __restrict__ s2) {
  int t = blockIdx.x * blockDim.x + threadIdx.x;  // t = bh*N + n
  if (t >= B * H * N) return;
  int h = (t / N) & 7;
  const _Float16* hr = hid + (size_t)t * COUT;
  const float* a0 = a + (size_t)h * 2 * COUT;
  const float* a1 = a0 + COUT;
  float r0 = 0.f, r1 = 0.f;
#pragma unroll 8
  for (int o = 0; o < COUT; ++o) {
    float v = (float)hr[o];
    r0 += v * a0[o];
    r1 += v * a1[o];
  }
  s1[t] = r0;
  s2[t] = r1;
}

// ---------------------------------------------------------------------------
// Fused masked-softmax attention: out = softmax(lrelu(s1+s2)+mask) @ hidden
// Block: 256 thr = 8 waves; wave w owns rows r0..r0+15 of one (b,h).
// Loop over 64-wide column chunks; V tile staged in LDS once per block.
__global__ __launch_bounds__(256) void k_attn(
    const _Float16* __restrict__ hidT,  // [B*H, COUT, N]
    const float* __restrict__ s1,       // [B*H, N]
    const float* __restrict__ s2,       // [B*H, N]
    const unsigned* __restrict__ mbits, // [B, N, N/32]
    const float* __restrict__ bias,     // [H, COUT]
    float* __restrict__ out) {          // [B, N, H*COUT]
  __shared__ v16h  ldsV[COUT * 4];      // [o][4 x 16 halfs] = 64 m's, 8 KB
  __shared__ float ldsS2[64];

  const int bh = blockIdx.x, b = bh >> 3, h = bh & 7;
  const int wave = threadIdx.x >> 5, lane = threadIdx.x & 31;
  const int n = lane & 15;
  const bool hi = lane >= 16;
  const int r0 = blockIdx.y * 128 + wave * 16;
  const int myrow = r0 + n;  // row handled by this lane (dup at lane, lane+16)

  const float s1v = s1[bh * N + myrow];
  const unsigned* mrow = mbits + (size_t)(b * N + myrow) * (N / 32);

  v8f acc[4] = {};
  float m_i = -3.0e38f, l_i = 0.f;

  for (int mc = 0; mc < N; mc += 64) {
    __syncthreads();
    {  // stage V chunk transposed: ldsV[o][0..63] = hidT[bh, o, mc..mc+63]
      const int o = threadIdx.x >> 2, q = threadIdx.x & 3;
      ldsV[o * 4 + q] =
          *(const v16h*)(hidT + (size_t)(bh * COUT + o) * N + mc + q * 16);
      if (threadIdx.x < 64) ldsS2[threadIdx.x] = s2[bh * N + mc + threadIdx.x];
    }
    __syncthreads();

#pragma unroll
    for (int sub = 0; sub < 2; ++sub) {
      const int kb = hi ? 16 : 0;
      const unsigned mw = mrow[(mc >> 5) + sub];

      // scores in A-fragment order: K = kb+{0..15} contiguous
      float p[16];
      float cmax = -3.0e38f;
#pragma unroll
      for (int i = 0; i < 8; ++i) {
        float sc0 = s1v + ldsS2[sub * 32 + kb + i];
        sc0 = sc0 >= 0.f ? sc0 : LRELU_ALPHA * sc0;
        if (!((mw >> (kb + i)) & 1u)) sc0 += NEG_BIG;
        float sc1 = s1v + ldsS2[sub * 32 + kb + 8 + i];
        sc1 = sc1 >= 0.f ? sc1 : LRELU_ALPHA * sc1;
        if (!((mw >> (kb + 8 + i)) & 1u)) sc1 += NEG_BIG;
        p[i] = sc0;
        p[8 + i] = sc1;
        cmax = fmaxf(cmax, fmaxf(sc0, sc1));
      }
      // combine with partner lane (same row, other K half)
      cmax = fmaxf(cmax, __shfl_xor(cmax, 16, 32));
      const float mnew  = fmaxf(m_i, cmax);
      const float alpha = __expf(m_i - mnew);
      float rsum = 0.f;
#pragma unroll
      for (int k = 0; k < 16; ++k) {
        p[k] = __expf(p[k] - mnew);
        rsum += p[k];
      }
      rsum += __shfl_xor(rsum, 16, 32);
      l_i = l_i * alpha + rsum;
      m_i = mnew;

      // rescale accumulators: C-row (hi?8+r:r)'s alpha lives at that lane id
#pragma unroll
      for (int r = 0; r < 8; ++r) {
        const float f = __shfl(alpha, hi ? (r + 8) : r, 32);
#pragma unroll
        for (int j = 0; j < 4; ++j) acc[j][r] *= f;
      }

      // P as f16 A-fragment (already in register order)
      v16h af;
#pragma unroll
      for (int k = 0; k < 16; ++k) af[k] = (_Float16)p[k];

      const int vq = sub * 2 + (hi ? 1 : 0);  // 16-half group within chunk
#pragma unroll
      for (int j = 0; j < 4; ++j) {
        const v16h bf = ldsV[(j * 16 + n) * 4 + vq];
        acc[j] = __builtin_amdgcn_wmma_f32_16x16x32_f16(
            false, af, false, bf, (short)0, acc[j], false, false);
      }
    }
  }

  // epilogue: normalize, +bias, ReLU, head-concat layout [B, N, H*COUT]
#pragma unroll
  for (int r = 0; r < 8; ++r) {
    const int m = r0 + (hi ? 8 + r : r);
    const float lr = __shfl(l_i, hi ? (r + 8) : r, 32);
    const float inv = 1.0f / lr;
#pragma unroll
    for (int j = 0; j < 4; ++j) {
      const int o = j * 16 + n;
      float v = acc[j][r] * inv + bias[h * COUT + o];
      out[(size_t)(b * N + m) * (H * COUT) + h * COUT + o] = v > 0.f ? v : 0.f;
    }
  }
}

// ---------------------------------------------------------------------------
extern "C" void kernel_launch(void* const* d_in, const int* in_sizes, int n_in,
                              void* d_out, int out_size, void* d_ws,
                              size_t ws_size, hipStream_t stream) {
  (void)in_sizes; (void)n_in; (void)out_size; (void)ws_size;
  const float* x    = (const float*)d_in[0];
  const int*   adj  = (const int*)d_in[1];
  const float* W    = (const float*)d_in[2];
  const float* a    = (const float*)d_in[3];
  const float* bias = (const float*)d_in[4];
  float* out = (float*)d_out;

  // workspace carve-out (~24 MB total), 256B aligned slices
  char* ws = (char*)d_ws;
  size_t off = 0;
  auto carve = [&](size_t bytes) -> void* {
    void* p = ws + off;
    off += (bytes + 255) & ~(size_t)255;
    return p;
  };
  _Float16* x16  = (_Float16*)carve((size_t)B * N * CIN * 2);        // 4 MB
  _Float16* w16  = (_Float16*)carve((size_t)H * COUT * CIN * 2);     // 256 KB
  _Float16* hid  = (_Float16*)carve((size_t)B * H * N * COUT * 2);   // 8 MB
  _Float16* hidT = (_Float16*)carve((size_t)B * H * N * COUT * 2);   // 8 MB
  float*    s1   = (float*)carve((size_t)B * H * N * 4);             // 256 KB
  float*    s2   = (float*)carve((size_t)B * H * N * 4);             // 256 KB
  unsigned* mb   = (unsigned*)carve((size_t)B * N * (N / 32) * 4);   // 2 MB

  const int nx = B * N * CIN;
  k_f32_to_f16<<<(nx + 255) / 256, 256, 0, stream>>>(x, x16, nx);
  const int nw = H * COUT * CIN;
  k_f32_to_f16<<<(nw + 255) / 256, 256, 0, stream>>>(W, w16, nw);

  const int nwords = B * N * (N / 32);
  k_pack_adj<<<(nwords * 32 + 255) / 256, 256, 0, stream>>>(adj, mb, nwords);

  k_hidden<<<dim3(B * H, N / 128), 256, 0, stream>>>(x16, w16, hid, hidT);
  k_scores<<<(B * H * N + 255) / 256, 256, 0, stream>>>(hid, a, s1, s2);
  k_attn<<<dim3(B * H, N / 128), 256, 0, stream>>>(hidT, s1, s2, mb, bias, out);
}
